// BTBlock_37099927502936
// MI455X (gfx1250) — matrix operand order repaired and checked by reading, not compiled, not run on hardware
//
#include <hip/hip_runtime.h>
#include <hip/hip_bf16.h>
#include <math.h>

// ---------------------------------------------------------------------------
// CDNA5 / gfx1250 implementation of the dual-branch shifted-window block.
// All dense linear algebra goes through v_wmma_f32_16x16x32_f16 (wave32).
// ---------------------------------------------------------------------------

typedef _Float16 half_t;
typedef __attribute__((ext_vector_type(16))) _Float16 v16h;
typedef __attribute__((ext_vector_type(8)))  float    v8f;

static constexpr long  TOKN  = 50176;   // B*H*W == windows*49
static constexpr int   HH    = 56;
static constexpr int   NSEQ  = 49;
static constexpr int   SHIFTC = 3;
static constexpr float SCALEC = 0.17677669529663687f; // 32^-0.5

enum { GF_GELU = 1, GF_ADD = 2 };

__device__ __forceinline__ float gelu_exact(float x) {
  return 0.5f * x * (1.0f + erff(x * 0.70710678118654752f));
}

// ------------------------- generic WMMA GEMM -------------------------------
// C[M,N] = act( (A[M,K] @ W[N,K]^T + bias[N]) * alpha )
// A, W are f16; optional f16 store (Ch) and/or f32 store/accumulate (Cf).
// Requires M%128==0, N%64==0, K%32==0 (true for every call below).
__global__ __launch_bounds__(256) void gemm_wmma_f16(
    const half_t* __restrict__ A, const half_t* __restrict__ W,
    const float* __restrict__ bias, half_t* __restrict__ Ch,
    float* __restrict__ Cf, int M, int N, int K, int flags, float alpha)
{
  __shared__ half_t As[128][40];   // row stride 80B (16B-aligned)
  __shared__ half_t Ws[64][40];

  const int tid  = threadIdx.x;
  const int lane = tid & 31;
  const int wv   = tid >> 5;
  const int wr   = wv >> 1;                 // wave row 0..3 (32 rows each)
  const int wc   = wv & 1;                  // wave col 0..1 (32 cols each)
  const long row0 = (long)blockIdx.x * 128;
  const long col0 = (long)blockIdx.y * 64;

  v8f acc[2][2] = {};

  const int ar = tid >> 1, aseg = (tid & 1) * 16;   // A tile: 16 halfs/thread
  const int wrw = tid >> 2, wseg = (tid & 3) * 8;   // W tile: 8 halfs/thread

  for (int k0 = 0; k0 < K; k0 += 32) {
    const half_t* ag = A + (row0 + ar) * (long)K + k0;
    const half_t* wg = W + (col0 + wrw) * (long)K + k0;
    *(float4*)(&As[ar][aseg])     = *(const float4*)(ag + aseg);
    *(float4*)(&As[ar][aseg + 8]) = *(const float4*)(ag + aseg + 8);
    *(float4*)(&Ws[wrw][wseg])    = *(const float4*)(wg + wseg);
    if (k0 + 32 < K) {                       // gfx1250 global_prefetch_b8
      __builtin_prefetch(ag + 32, 0, 1);
      __builtin_prefetch(wg + 32, 0, 1);
    }
    __syncthreads();

    // Fragment layouts per CDNA5 ISA 7.12.2 (16-bit A 16x32, B 32x16).
    const int m  = lane & 15;
    const int kb = (lane >> 4) << 3;   // A: K base 0 / 8
    const int ks = (lane >> 4) << 4;   // K base 0 / 16 (unused; B loads use kb)
    v16h af[2], bf[2];
#pragma unroll
    for (int i = 0; i < 2; ++i) {
      const half_t* p = &As[wr * 32 + i * 16 + m][0];
      v16h a;
#pragma unroll
      for (int e = 0; e < 8; ++e) { a[e] = p[kb + e]; a[8 + e] = p[16 + kb + e]; }
      af[i] = a;
    }
#pragma unroll
    for (int j = 0; j < 2; ++j) {
      const half_t* p = &Ws[wc * 32 + j * 16 + m][0];  // column = lane&15
      v16h b;
#pragma unroll
      for (int e = 0; e < 8; ++e) { b[e] = p[kb + e]; b[8 + e] = p[16 + kb + e]; }
      bf[j] = b;
    }
#pragma unroll
    for (int i = 0; i < 2; ++i)
#pragma unroll
      for (int j = 0; j < 2; ++j)
        acc[i][j] = __builtin_amdgcn_wmma_f32_16x16x32_f16(
            false, af[i], false, bf[j], (short)0, acc[i][j], false, false);
    __syncthreads();
  }

  const int rof = (lane >> 4) << 3;
  const int cof = lane & 15;
#pragma unroll
  for (int i = 0; i < 2; ++i)
#pragma unroll
    for (int j = 0; j < 2; ++j)
#pragma unroll
      for (int r = 0; r < 8; ++r) {
        long row = row0 + wr * 32 + i * 16 + r + rof;
        long col = col0 + wc * 32 + j * 16 + cof;
        float v = (acc[i][j][r] + bias[col]) * alpha;
        if (flags & GF_GELU) v = gelu_exact(v);
        long idx = row * (long)N + col;
        if (Ch) Ch[idx] = (half_t)v;
        if (Cf) { if (flags & GF_ADD) Cf[idx] += v; else Cf[idx] = v; }
      }
}

// ------------------------- attention (per window, head) --------------------
// S = qn·kn^T (+bias+mask), softmax rows, O = P·V, out = O - qsub_scale*q.
__global__ __launch_bounds__(256) void attn_wmma(
    const half_t* __restrict__ qn, const half_t* __restrict__ kn,
    const half_t* __restrict__ kvo /* [tok][512], v at +256 */,
    const half_t* __restrict__ qsc, const float* __restrict__ biasf,
    const float* __restrict__ maskf, half_t* __restrict__ out, float qsub_scale)
{
  __shared__ half_t Qs[64][40];
  __shared__ half_t Ks[64][40];
  __shared__ half_t Vs[64][40];
  __shared__ float  Sf[64][68];
  __shared__ half_t Ps[64][72];

  const int w = blockIdx.x;
  const int h = blockIdx.y;
  const int tid = threadIdx.x;
  const int lane = tid & 31;
  const int wv = tid >> 5;
  const long tb = (long)w * NSEQ;

  for (int i = tid; i < 64 * 32; i += 256) {
    int n = i >> 5, d = i & 31;
    half_t qv = (half_t)0.f, kv = (half_t)0.f, vv = (half_t)0.f;
    if (n < NSEQ) {
      long t = tb + n;
      qv = qn[t * 256 + h * 32 + d];
      kv = kn[t * 256 + h * 32 + d];
      vv = kvo[t * 512 + 256 + h * 32 + d];
    }
    Qs[n][d] = qv; Ks[n][d] = kv; Vs[n][d] = vv;
  }
  __syncthreads();

  const float* bh = biasf + (long)h * NSEQ * NSEQ;
  const float* mw = maskf + (long)(w & 63) * NSEQ * NSEQ;
  const int m   = lane & 15;
  const int kb  = (lane >> 4) << 3;
  const int ks  = (lane >> 4) << 4;
  const int rof = (lane >> 4) << 3;
  const int cof = lane & 15;

  // S: 4x4 tiles of 16x16, K=32 (head dim) -> one WMMA per tile, 2 tiles/wave.
#pragma unroll
  for (int tt = 0; tt < 2; ++tt) {
    int t = wv + tt * 8;
    int tm = t >> 2, tn = t & 3;
    v16h a, b;
    const half_t* pa = &Qs[tm * 16 + m][0];
#pragma unroll
    for (int e = 0; e < 8; ++e) { a[e] = pa[kb + e]; a[8 + e] = pa[16 + kb + e]; }
    const half_t* pb = &Ks[tn * 16 + cof][0];          // B(k,n) = K[n][k]
#pragma unroll
    for (int e = 0; e < 8; ++e) { b[e] = pb[kb + e]; b[8 + e] = pb[16 + kb + e]; }
    v8f c = {};
    c = __builtin_amdgcn_wmma_f32_16x16x32_f16(false, a, false, b, (short)0, c,
                                               false, false);
#pragma unroll
    for (int r = 0; r < 8; ++r) {
      int row = tm * 16 + r + rof;
      int col = tn * 16 + cof;
      float v;
      if (row < NSEQ && col < NSEQ)
        v = c[r] + bh[row * NSEQ + col] + mw[row * NSEQ + col];
      else
        v = -1e30f;                                     // kill padded columns
      Sf[row][col] = v;
    }
  }
  __syncthreads();

  if (tid < 64) {                                       // row softmax
    float mx = -1e30f;
    for (int c2 = 0; c2 < 64; ++c2) mx = fmaxf(mx, Sf[tid][c2]);
    float s = 0.f;
    for (int c2 = 0; c2 < 64; ++c2) s += __expf(Sf[tid][c2] - mx);
    float inv = 1.f / s;
    for (int c2 = 0; c2 < 64; ++c2)
      Ps[tid][c2] = (half_t)(__expf(Sf[tid][c2] - mx) * inv);
  }
  __syncthreads();

  // O = P(64x64) @ V(64x32): 4x2 tiles, one per wave, K=64 -> 2 chained WMMAs.
  const int tm = wv >> 1, tn = wv & 1;
  v8f c = {};
#pragma unroll
  for (int kh = 0; kh < 2; ++kh) {
    v16h a, b;
    const half_t* pa = &Ps[tm * 16 + m][kh * 32];
#pragma unroll
    for (int e = 0; e < 8; ++e) { a[e] = pa[kb + e]; a[8 + e] = pa[16 + kb + e]; }
#pragma unroll
    for (int e = 0; e < 8; ++e) { b[e] = Vs[kh * 32 + kb + e][tn * 16 + cof]; b[8 + e] = Vs[kh * 32 + 16 + kb + e][tn * 16 + cof]; }
    c = __builtin_amdgcn_wmma_f32_16x16x32_f16(false, a, false, b, (short)0, c,
                                               false, false);
  }
#pragma unroll
  for (int r = 0; r < 8; ++r) {
    int row = tm * 16 + r + rof;
    int col = tn * 16 + cof;
    if (row < NSEQ) {
      long idx = (tb + row) * 256 + h * 32 + col;
      out[idx] = (half_t)(c[r] - qsub_scale * (float)qsc[idx]);
    }
  }
}

// ------------------------- elementwise helpers -----------------------------
__device__ __forceinline__ float block_sum_256(float v, float* red) {
#pragma unroll
  for (int o = 16; o > 0; o >>= 1) v += __shfl_down(v, o, 32);
  __syncthreads();
  if ((threadIdx.x & 31) == 0) red[threadIdx.x >> 5] = v;
  __syncthreads();
  float s = 0.f;
#pragma unroll
  for (int i = 0; i < 8; ++i) s += red[i];
  return s;
}

// LN(norm1) + roll(-3,-3) + window partition, writes xw1, xw2, |xw2-xw1| f16.
__global__ __launch_bounds__(256) void ln_shift_window(
    const float* __restrict__ x1, const float* __restrict__ x2,
    const float* __restrict__ g, const float* __restrict__ bta,
    half_t* __restrict__ xw1, half_t* __restrict__ xw2,
    half_t* __restrict__ qabs)
{
  __shared__ float red[8];
  const long t = blockIdx.x;                   // windowed token
  const int bw = (int)(t / NSEQ), n = (int)(t % NSEQ);
  const int bb = bw >> 6, wi = bw & 63;
  const int wh = wi >> 3, ww = wi & 7;
  const int ih = n / 7, iw = n % 7;
  const int hs = (wh * 7 + ih + SHIFTC) % HH;
  const int wsx = (ww * 7 + iw + SHIFTC) % HH;
  const long src = ((long)bb * (HH * HH) + hs * HH + wsx) * 256;
  const int cdx = threadIdx.x;
  float v1 = x1[src + cdx], v2 = x2[src + cdx];
  float s1 = block_sum_256(v1, red);
  float q1 = block_sum_256(v1 * v1, red);
  float s2 = block_sum_256(v2, red);
  float q2 = block_sum_256(v2 * v2, red);
  float m1 = s1 * (1.f / 256.f), m2 = s2 * (1.f / 256.f);
  float r1 = rsqrtf(fmaxf(q1 * (1.f / 256.f) - m1 * m1, 0.f) + 1e-5f);
  float r2 = rsqrtf(fmaxf(q2 * (1.f / 256.f) - m2 * m2, 0.f) + 1e-5f);
  float a1 = (v1 - m1) * r1 * g[cdx] + bta[cdx];
  float a2 = (v2 - m2) * r2 * g[cdx] + bta[cdx];
  long o = t * 256 + cdx;
  xw1[o] = (half_t)a1;
  xw2[o] = (half_t)a2;
  qabs[o] = (half_t)fabsf(a2 - a1);
}

__global__ __launch_bounds__(256) void ln_plain(
    const float* __restrict__ x, const float* __restrict__ g,
    const float* __restrict__ bta, half_t* __restrict__ out)
{
  __shared__ float red[8];
  const long t = blockIdx.x;
  const int cdx = threadIdx.x;
  float v = x[t * 256 + cdx];
  float s = block_sum_256(v, red);
  float q = block_sum_256(v * v, red);
  float m = s * (1.f / 256.f);
  float r = rsqrtf(fmaxf(q * (1.f / 256.f) - m * m, 0.f) + 1e-5f);
  out[t * 256 + cdx] = (half_t)((v - m) * r * g[cdx] + bta[cdx]);
}

// L2 normalize across the HEADS axis (torch F.normalize dim=1).
__global__ __launch_bounds__(256) void l2norm_heads(
    const half_t* __restrict__ in, int stride, half_t* __restrict__ out,
    long tokens)
{
  long i = (long)blockIdx.x * 256 + threadIdx.x;
  if (i >= tokens * 32) return;
  long t = i >> 5;
  int d = (int)(i & 31);
  float s = 0.f;
#pragma unroll
  for (int h = 0; h < 8; ++h) {
    float v = (float)in[t * stride + h * 32 + d];
    s += v * v;
  }
  float inv = 1.f / fmaxf(sqrtf(s), 1e-12f);
#pragma unroll
  for (int h = 0; h < 8; ++h)
    out[t * 256 + h * 32 + d] =
        (half_t)((float)in[t * stride + h * 32 + d] * inv);
}

__global__ __launch_bounds__(256) void dwconv_gelu(
    const float* __restrict__ x, const float* __restrict__ dw,
    const float* __restrict__ db, half_t* __restrict__ out)
{
  long i = (long)blockIdx.x * 256 + threadIdx.x;
  int cdx = (int)(i & 255);
  long t = i >> 8;
  int bb = (int)(t / 3136);
  int hw = (int)(t % 3136);
  int hh = hw / HH, wwp = hw % HH;
  float s = db[cdx];
#pragma unroll
  for (int kh = 0; kh < 3; ++kh)
#pragma unroll
    for (int kw = 0; kw < 3; ++kw) {
      int yy = hh + kh - 1, xx = wwp + kw - 1;
      if (yy >= 0 && yy < HH && xx >= 0 && xx < HH)
        s += x[((long)bb * 3136 + yy * HH + xx) * 256 + cdx] *
             dw[(kh * 3 + kw) * 256 + cdx];
    }
  out[i] = (half_t)gelu_exact(s);
}

// window reverse + roll(+3,+3) + residual + LFE sum -> d_out (both branches)
__global__ __launch_bounds__(256) void combine_out(
    const float* __restrict__ x1, const float* __restrict__ x2,
    const float* __restrict__ y1, const float* __restrict__ y2,
    const float* __restrict__ c1, const float* __restrict__ c2,
    float* __restrict__ o1, float* __restrict__ o2)
{
  long i = (long)blockIdx.x * 256 + threadIdx.x;
  int cdx = (int)(i & 255);
  long t = i >> 8;
  int bb = (int)(t / 3136);
  int hw = (int)(t % 3136);
  int hh = hw / HH, wwp = hw % HH;
  int hs2 = (hh - SHIFTC + HH) % HH;
  int ws2 = (wwp - SHIFTC + HH) % HH;
  int wh = hs2 / 7, ih = hs2 % 7;
  int wc = ws2 / 7, iw = ws2 % 7;
  long tw = ((long)(bb * 64 + wh * 8 + wc)) * NSEQ + ih * 7 + iw;
  o1[i] = x1[i] + y1[tw * 256 + cdx] + c1[i];
  o2[i] = x2[i] + y2[tw * 256 + cdx] + c2[i];
}

__global__ __launch_bounds__(256) void cvt_f16(const float* __restrict__ s,
                                               half_t* __restrict__ d, long n)
{
  long i = (long)blockIdx.x * 256 + threadIdx.x;
  if (i < n) d[i] = (half_t)s[i];
}

// pw conv weight [1,1,in=256,out=256] -> [out][in] f16 (GEMM W layout)
__global__ __launch_bounds__(256) void cvt_f16_T256(const float* __restrict__ s,
                                                    half_t* __restrict__ d)
{
  int i = blockIdx.x * 256 + threadIdx.x;
  int r = i >> 8, c = i & 255;
  d[c * 256 + r] = (half_t)s[i];
}

// bias_full[h][m][n] = bias_table[rel_idx[m][n]][h]
__global__ __launch_bounds__(256) void bias_gather(
    const float* __restrict__ table, const int* __restrict__ rel,
    float* __restrict__ out)
{
  int i = blockIdx.x * 256 + threadIdx.x;
  if (i >= 8 * NSEQ * NSEQ) return;
  int h = i / (NSEQ * NSEQ);
  int mn = i % (NSEQ * NSEQ);
  out[i] = table[rel[mn] * 8 + h];
}

// ---------------------------------------------------------------------------
extern "C" void kernel_launch(void* const* d_in, const int* in_sizes, int n_in,
                              void* d_out, int out_size, void* d_ws,
                              size_t ws_size, hipStream_t stream)
{
  (void)in_sizes; (void)n_in; (void)out_size; (void)ws_size;
  const float* x1 = (const float*)d_in[0];
  const float* x2 = (const float*)d_in[1];
  const float* norm1_g = (const float*)d_in[2];
  const float* norm1_b = (const float*)d_in[3];
  const float* norm2_g = (const float*)d_in[4];
  const float* norm2_b = (const float*)d_in[5];
  const float* kv1_w = (const float*)d_in[6];
  const float* kv1_b = (const float*)d_in[7];
  const float* kv2_w = (const float*)d_in[8];
  const float* kv2_b = (const float*)d_in[9];
  const float* q_w  = (const float*)d_in[10];
  const float* q_b  = (const float*)d_in[11];
  const float* bias_table = (const float*)d_in[12];
  const float* proj1_w = (const float*)d_in[13];
  const float* proj1_b = (const float*)d_in[14];
  const float* proj2_w = (const float*)d_in[15];
  const float* proj2_b = (const float*)d_in[16];
  const float* m1f1_w = (const float*)d_in[17];
  const float* m1f1_b = (const float*)d_in[18];
  const float* m1f2_w = (const float*)d_in[19];
  const float* m1f2_b = (const float*)d_in[20];
  const float* m2f1_w = (const float*)d_in[21];
  const float* m2f1_b = (const float*)d_in[22];
  const float* m2f2_w = (const float*)d_in[23];
  const float* m2f2_b = (const float*)d_in[24];
  const float* dw1_w = (const float*)d_in[25];
  const float* dw1_b = (const float*)d_in[26];
  const float* pw1_w = (const float*)d_in[27];
  const float* pw1_b = (const float*)d_in[28];
  const float* dw2_w = (const float*)d_in[29];
  const float* dw2_b = (const float*)d_in[30];
  const float* pw2_w = (const float*)d_in[31];
  const float* pw2_b = (const float*)d_in[32];
  const int* rel_idx = (const int*)d_in[33];
  const float* attn_mask = (const float*)d_in[34];

  char* ws = (char*)d_ws;
  size_t off = 0;
  auto alloc = [&](size_t bytes) -> void* {
    void* p = ws + off;
    off += (bytes + 255) & ~(size_t)255;
    return p;
  };

  half_t* Wkv1 = (half_t*)alloc(512 * 256 * 2);
  half_t* Wkv2 = (half_t*)alloc(512 * 256 * 2);
  half_t* Wq   = (half_t*)alloc(256 * 256 * 2);
  half_t* Wp1  = (half_t*)alloc(256 * 256 * 2);
  half_t* Wp2  = (half_t*)alloc(256 * 256 * 2);
  half_t* Wf11 = (half_t*)alloc(1024 * 256 * 2);
  half_t* Wf12 = (half_t*)alloc(256 * 1024 * 2);
  half_t* Wf21 = (half_t*)alloc(1024 * 256 * 2);
  half_t* Wf22 = (half_t*)alloc(256 * 1024 * 2);
  half_t* Wpw1 = (half_t*)alloc(256 * 256 * 2);
  half_t* Wpw2 = (half_t*)alloc(256 * 256 * 2);
  float*  biasf = (float*)alloc(8 * NSEQ * NSEQ * 4);
  half_t* xw1  = (half_t*)alloc(TOKN * 256 * 2);
  half_t* xw2  = (half_t*)alloc(TOKN * 256 * 2);
  half_t* qabs = (half_t*)alloc(TOKN * 256 * 2);
  half_t* kv1o = (half_t*)alloc(TOKN * 1024 * 2); // [tok][512] + second half
  half_t* kv2o = kv1o + TOKN * 512;
  half_t* hidden = kv1o;                           // MLP reuse (TOKN x 1024)
  half_t* qo  = (half_t*)alloc(TOKN * 256 * 2);
  half_t* qnb = (half_t*)alloc(TOKN * 256 * 2);
  half_t* kn1 = (half_t*)alloc(TOKN * 256 * 2);
  half_t* kn2 = (half_t*)alloc(TOKN * 256 * 2);
  half_t* o1b = (half_t*)alloc(TOKN * 256 * 2);
  half_t* o2b = (half_t*)alloc(TOKN * 256 * 2);
  float* y1 = (float*)alloc(TOKN * 256 * 4);
  float* y2 = (float*)alloc(TOKN * 256 * 4);
  float* c1 = (float*)alloc(TOKN * 256 * 4);
  float* c2 = (float*)alloc(TOKN * 256 * 4);
  half_t* g1  = qabs;  // free after q GEMM
  half_t* g2  = qnb;   // free after attention
  half_t* lnb = xw1;   // free after kv1 GEMM

  float* out1 = (float*)d_out;
  float* out2 = out1 + TOKN * 256;

  dim3 blk(256);
  auto cdiv = [](long a, long b) { return (unsigned)((a + b - 1) / b); };

  // --- weight prep (tiny; stays resident in L2) ---
  cvt_f16<<<cdiv(512 * 256, 256), blk, 0, stream>>>(kv1_w, Wkv1, 512 * 256);
  cvt_f16<<<cdiv(512 * 256, 256), blk, 0, stream>>>(kv2_w, Wkv2, 512 * 256);
  cvt_f16<<<cdiv(256 * 256, 256), blk, 0, stream>>>(q_w, Wq, 256 * 256);
  cvt_f16<<<cdiv(256 * 256, 256), blk, 0, stream>>>(proj1_w, Wp1, 256 * 256);
  cvt_f16<<<cdiv(256 * 256, 256), blk, 0, stream>>>(proj2_w, Wp2, 256 * 256);
  cvt_f16<<<cdiv(1024 * 256, 256), blk, 0, stream>>>(m1f1_w, Wf11, 1024 * 256);
  cvt_f16<<<cdiv(256 * 1024, 256), blk, 0, stream>>>(m1f2_w, Wf12, 256 * 1024);
  cvt_f16<<<cdiv(1024 * 256, 256), blk, 0, stream>>>(m2f1_w, Wf21, 1024 * 256);
  cvt_f16<<<cdiv(256 * 1024, 256), blk, 0, stream>>>(m2f2_w, Wf22, 256 * 1024);
  cvt_f16_T256<<<256, blk, 0, stream>>>(pw1_w, Wpw1);
  cvt_f16_T256<<<256, blk, 0, stream>>>(pw2_w, Wpw2);
  bias_gather<<<cdiv(8 * NSEQ * NSEQ, 256), blk, 0, stream>>>(bias_table,
                                                              rel_idx, biasf);

  // --- LN + shift + window partition ---
  ln_shift_window<<<(unsigned)TOKN, blk, 0, stream>>>(
      x1, x2, norm1_g, norm1_b, xw1, xw2, qabs);

  // --- QKV projections (WMMA) ---
  gemm_wmma_f16<<<dim3(392, 8), blk, 0, stream>>>(
      xw1, Wkv1, kv1_b, kv1o, nullptr, (int)TOKN, 512, 256, 0, 1.f);
  gemm_wmma_f16<<<dim3(392, 8), blk, 0, stream>>>(
      xw2, Wkv2, kv2_b, kv2o, nullptr, (int)TOKN, 512, 256, 0, 1.f);
  gemm_wmma_f16<<<dim3(392, 4), blk, 0, stream>>>(
      qabs, Wq, q_b, qo, nullptr, (int)TOKN, 256, 256, 0, SCALEC);

  // --- cross-head L2 normalize q, k1, k2 ---
  l2norm_heads<<<cdiv(TOKN * 32, 256), blk, 0, stream>>>(qo, 256, qnb, TOKN);
  l2norm_heads<<<cdiv(TOKN * 32, 256), blk, 0, stream>>>(kv2o, 512, kn2, TOKN);
  l2norm_heads<<<cdiv(TOKN * 32, 256), blk, 0, stream>>>(kv1o, 512, kn1, TOKN);

  // --- attention, both branches (WMMA) ---
  attn_wmma<<<dim3(1024, 8), blk, 0, stream>>>(qnb, kn2, kv2o, qo, biasf,
                                               attn_mask, o1b, 1.f);
  attn_wmma<<<dim3(1024, 8), blk, 0, stream>>>(qnb, kn1, kv1o, qo, biasf,
                                               attn_mask, o2b, SCALEC);

  // --- output projections ---
  gemm_wmma_f16<<<dim3(392, 4), blk, 0, stream>>>(
      o1b, Wp1, proj1_b, nullptr, y1, (int)TOKN, 256, 256, 0, 1.f);
  gemm_wmma_f16<<<dim3(392, 4), blk, 0, stream>>>(
      o2b, Wp2, proj2_b, nullptr, y2, (int)TOKN, 256, 256, 0, 1.f);

  // --- LFE: depthwise+GELU then pointwise (WMMA) ---
  dwconv_gelu<<<(unsigned)TOKN, blk, 0, stream>>>(x1, dw1_w, dw1_b, g1);
  dwconv_gelu<<<(unsigned)TOKN, blk, 0, stream>>>(x2, dw2_w, dw2_b, g2);
  gemm_wmma_f16<<<dim3(392, 4), blk, 0, stream>>>(
      g1, Wpw1, pw1_b, nullptr, c1, (int)TOKN, 256, 256, 0, 1.f);
  gemm_wmma_f16<<<dim3(392, 4), blk, 0, stream>>>(
      g2, Wpw2, pw2_b, nullptr, c2, (int)TOKN, 256, 256, 0, 1.f);

  // --- window reverse + roll + residual + LFE sum -> d_out ---
  combine_out<<<(unsigned)TOKN, blk, 0, stream>>>(x1, x2, y1, y2, c1, c2, out1,
                                                  out2);

  // --- MLP branch 1 (fc1 fused GELU, fc2 fused residual add) ---
  ln_plain<<<(unsigned)TOKN, blk, 0, stream>>>(out1, norm2_g, norm2_b, lnb);
  gemm_wmma_f16<<<dim3(392, 16), blk, 0, stream>>>(
      lnb, Wf11, m1f1_b, hidden, nullptr, (int)TOKN, 1024, 256, GF_GELU, 1.f);
  gemm_wmma_f16<<<dim3(392, 4), blk, 0, stream>>>(
      hidden, Wf12, m1f2_b, nullptr, out1, (int)TOKN, 256, 1024, GF_ADD, 1.f);

  // --- MLP branch 2 ---
  ln_plain<<<(unsigned)TOKN, blk, 0, stream>>>(out2, norm2_g, norm2_b, lnb);
  gemm_wmma_f16<<<dim3(392, 16), blk, 0, stream>>>(
      lnb, Wf21, m2f1_b, hidden, nullptr, (int)TOKN, 1024, 256, GF_GELU, 1.f);
  gemm_wmma_f16<<<dim3(392, 4), blk, 0, stream>>>(
      hidden, Wf22, m2f2_b, nullptr, out2, (int)TOKN, 256, 1024, GF_ADD, 1.f);
}
